// RetroactiveModel_35270271435188
// MI455X (gfx1250) — matrix operand order repaired and checked by reading, not compiled, not run on hardware
//
#include <hip/hip_runtime.h>
#include <hip/hip_bf16.h>
#include <math.h>

#define V_SZ 32000
#define H_SZ 1024
#define B_SZ 16
#define T_SZ 2048
#define MT (B_SZ * T_SZ)   // 32768 token rows
#define N1 (2 * H_SZ)      // 2048
#define SLOTS 8
#define K1_SLOTS 6
#define K2_SLOTS 2

typedef __attribute__((ext_vector_type(16))) __bf16 v16bf;
typedef __attribute__((ext_vector_type(8)))  __bf16 v8bf;
typedef __attribute__((ext_vector_type(8)))  float  v8f;

__device__ __forceinline__ unsigned short f2bf(float f) {
  unsigned int x = __float_as_uint(f);
  x += 0x7FFFu + ((x >> 16) & 1u);   // round-to-nearest-even
  return (unsigned short)(x >> 16);
}

__device__ __forceinline__ float blk_sum(float v, float* sbuf) {
  const int tid = threadIdx.x;
  sbuf[tid] = v; __syncthreads();
  for (int s = 128; s > 0; s >>= 1) {
    if (tid < s) sbuf[tid] += sbuf[tid + s];
    __syncthreads();
  }
  float r = sbuf[0]; __syncthreads();
  return r;
}

// ---------------------------------------------------------------- embedding
__global__ __launch_bounds__(256) void k_embed_gather(
    const int* __restrict__ seq, const float* __restrict__ embed,
    float* __restrict__ e_f32, unsigned short* __restrict__ e_bf16) {
  const int row = blockIdx.x;
  const int tok = seq[row];
  const float* src = embed + (size_t)tok * H_SZ;
  float* df = e_f32 + (size_t)row * H_SZ;
  unsigned short* db = e_bf16 + (size_t)row * H_SZ;
  for (int c = threadIdx.x; c < H_SZ; c += 256) {
    float v = src[c];
    df[c] = v;
    db[c] = f2bf(v);
  }
}

// ------------------------------------------- weight transpose + bf16 convert
__global__ __launch_bounds__(256) void k_transpose_bf16(
    const float* __restrict__ W, unsigned short* __restrict__ WT, int K, int N) {
  size_t idx = (size_t)blockIdx.x * 256 + threadIdx.x;
  if (idx >= (size_t)K * N) return;
  int k = (int)(idx / N);
  int n = (int)(idx % N);
  WT[(size_t)n * K + k] = f2bf(W[idx]);
}

// ---------------------------------------------------------------- WMMA GEMM
// C[M,N] = A[M,K] (bf16, row-major) * BT[N,K]^T (bf16) + bias[N]
// mode 0: ReLU, store bf16.  mode 1: add addend[M,N], store f32.
__global__ __launch_bounds__(256) void k_gemm_bf16(
    const unsigned short* __restrict__ A,
    const unsigned short* __restrict__ BT,
    const float* __restrict__ bias,
    const float* __restrict__ addend,
    unsigned short* __restrict__ Cbf,
    float* __restrict__ Cf,
    int N, int K, int mode)
{
  const int lane = threadIdx.x & 31;
  const int wave = threadIdx.x >> 5;
  const int wr = wave >> 1;                 // 0..3
  const int wc = wave & 1;                  // 0..1
  const int mW = blockIdx.x * 128 + wr * 32;
  const int nW = blockIdx.y * 128 + wc * 64;
  const int lm = lane & 15;
  const int kh = lane >> 4;

  v8f acc[2][4] = {};

  const unsigned short* aptr[2];
  aptr[0] = A + (size_t)(mW + lm) * K;
  aptr[1] = A + (size_t)(mW + 16 + lm) * K;
  const unsigned short* bptr[4];
#pragma unroll
  for (int j = 0; j < 4; ++j)
    bptr[j] = BT + (size_t)(nW + j * 16 + lm) * K;

  for (int k0 = 0; k0 < K; k0 += 32) {
    v16bf af[2];
#pragma unroll
    for (int i = 0; i < 2; ++i) {
      // A 16x32 bf16 layout: lane holds row lm; K = kh*16 + e
      v8bf lo = *(const v8bf*)(aptr[i] + k0 + kh * 16);
      v8bf hi = *(const v8bf*)(aptr[i] + k0 + kh * 16 + 8);
#pragma unroll
      for (int e = 0; e < 8; ++e) { af[i][e] = lo[e]; af[i][8 + e] = hi[e]; }
    }
    v16bf bfr[4];
#pragma unroll
    for (int j = 0; j < 4; ++j)   // B 32x16: lane holds col lm, K = kh*16 + e
      bfr[j] = *(const v16bf*)(bptr[j] + k0 + kh * 16);

#pragma unroll
    for (int i = 0; i < 2; ++i)
#pragma unroll
      for (int j = 0; j < 4; ++j)
        acc[i][j] = __builtin_amdgcn_wmma_f32_16x16x32_bf16(
            false, af[i], false, bfr[j], (short)0, acc[i][j], false, false);
  }

  // C layout: elem r -> row (r + kh*8), col lm within 16x16 tile
#pragma unroll
  for (int i = 0; i < 2; ++i) {
#pragma unroll
    for (int j = 0; j < 4; ++j) {
      const int nCol = nW + j * 16 + lm;
      const int mBase = mW + i * 16 + kh * 8;
      const float bv = bias[nCol];
#pragma unroll
      for (int r = 0; r < 8; ++r) {
        const size_t off = (size_t)(mBase + r) * N + nCol;
        float v = acc[i][j][r] + bv;
        if (mode == 0) {
          v = fmaxf(v, 0.0f);
          Cbf[off] = f2bf(v);
        } else {
          Cf[off] = v + addend[off];
        }
      }
    }
  }
}

// -------------------------------------------- LayerNorm + fused gate dot-prods
__global__ __launch_bounds__(256) void k_ln_gates(
    float* __restrict__ h, const float* __restrict__ g, const float* __restrict__ b,
    const float* __restrict__ g1w, const float* __restrict__ g1b,
    const float* __restrict__ g2w,
    float* __restrict__ gate1, float* __restrict__ dotg2)
{
  __shared__ float sbuf[256];
  const int tid = threadIdx.x;
  float* hr = h + (size_t)blockIdx.x * H_SZ;
  float x[4];
#pragma unroll
  for (int u = 0; u < 4; ++u) x[u] = hr[tid + u * 256];
  float s = x[0] + x[1] + x[2] + x[3];
  const float mean = blk_sum(s, sbuf) * (1.0f / H_SZ);
  float vs = 0.f;
#pragma unroll
  for (int u = 0; u < 4; ++u) { float d = x[u] - mean; vs += d * d; }
  const float var = blk_sum(vs, sbuf) * (1.0f / H_SZ);
  const float rinv = rsqrtf(var + 1e-5f);
  float d1 = 0.f, d2 = 0.f;
#pragma unroll
  for (int u = 0; u < 4; ++u) {
    const int c = tid + u * 256;
    const float y = (x[u] - mean) * rinv * g[c] + b[c];
    hr[c] = y;
    d1 += y * g1w[c];
    d2 += y * g2w[c];
  }
  const float D1 = blk_sum(d1, sbuf);
  const float D2 = blk_sum(d2, sbuf);
  if (tid == 0) {
    gate1[blockIdx.x] = 1.0f / (1.0f + expf(-(D1 + g1b[0])));
    dotg2[blockIdx.x] = D2;
  }
}

// ------------------------------------------ pass-1 top-6 + ctx1 mean + c2 scalar
__global__ __launch_bounds__(256) void k_select1(
    const float* __restrict__ gate1, const float* __restrict__ h,
    const float* __restrict__ g2w,
    float* __restrict__ ctxmean, float* __restrict__ c2, int* __restrict__ idx_all)
{
  __shared__ float sv[256];
  __shared__ int si[256];
  __shared__ unsigned char taken[T_SZ];
  __shared__ int chosen[K1_SLOTS];
  const int tid = threadIdx.x;
  const int bb = blockIdx.x;
  const float* gb = gate1 + (size_t)bb * T_SZ;
  for (int t = tid; t < T_SZ; t += 256) taken[t] = 0;
  __syncthreads();
  for (int s = 0; s < K1_SLOTS; ++s) {
    float best = -1e30f; int bi = T_SZ;
    for (int t = tid; t < T_SZ; t += 256) {
      if (!taken[t]) {
        float v = gb[t];
        if (v > best) { best = v; bi = t; }
      }
    }
    sv[tid] = best; si[tid] = bi; __syncthreads();
    for (int w = 128; w > 0; w >>= 1) {
      if (tid < w) {
        if (sv[tid + w] > sv[tid] ||
            (sv[tid + w] == sv[tid] && si[tid + w] < si[tid])) {
          sv[tid] = sv[tid + w]; si[tid] = si[tid + w];
        }
      }
      __syncthreads();
    }
    if (tid == 0) { chosen[s] = si[0]; taken[si[0]] = 1; }
    __syncthreads();
  }
  for (int c = tid; c < H_SZ; c += 256) {
    float accv = 0.f;
#pragma unroll
    for (int s = 0; s < K1_SLOTS; ++s)
      accv += h[((size_t)bb * T_SZ + chosen[s]) * H_SZ + c];
    ctxmean[(size_t)bb * H_SZ + c] = accv * (1.0f / K1_SLOTS);
  }
  __syncthreads();
  float p = 0.f;
  for (int c = tid; c < H_SZ; c += 256)
    p += ctxmean[(size_t)bb * H_SZ + c] * g2w[H_SZ + c];
  float tot = blk_sum(p, sv);
  if (tid == 0) c2[bb] = tot;
  if (tid < K1_SLOTS) idx_all[bb * SLOTS + tid] = chosen[tid];
}

// --------------------------------------------------------- pass-2 top-2 (masked)
__global__ __launch_bounds__(256) void k_select2(
    const float* __restrict__ dotg2, const float* __restrict__ c2,
    const float* __restrict__ g2b, int* __restrict__ idx_all)
{
  __shared__ float sv[256];
  __shared__ int si[256];
  __shared__ unsigned char taken[T_SZ];
  const int tid = threadIdx.x;
  const int bb = blockIdx.x;
  for (int t = tid; t < T_SZ; t += 256) taken[t] = 0;
  __syncthreads();
  if (tid < K1_SLOTS) taken[idx_all[bb * SLOTS + tid]] = 1;
  __syncthreads();
  const float cc = c2[bb] + g2b[0];
  const float* db = dotg2 + (size_t)bb * T_SZ;
  for (int s = 0; s < K2_SLOTS; ++s) {
    float best = -1e30f; int bi = T_SZ;
    for (int t = tid; t < T_SZ; t += 256) {
      if (!taken[t]) {
        float v = 1.0f / (1.0f + expf(-(db[t] + cc)));
        if (v > best) { best = v; bi = t; }
      }
    }
    sv[tid] = best; si[tid] = bi; __syncthreads();
    for (int w = 128; w > 0; w >>= 1) {
      if (tid < w) {
        if (sv[tid + w] > sv[tid] ||
            (sv[tid + w] == sv[tid] && si[tid + w] < si[tid])) {
          sv[tid] = sv[tid + w]; si[tid] = si[tid + w];
        }
      }
      __syncthreads();
    }
    if (tid == 0) { idx_all[bb * SLOTS + K1_SLOTS + s] = si[0]; taken[si[0]] = 1; }
    __syncthreads();
  }
}

// -------------------------------------------------- q = h[:, -1, :] @ q_w + q_b
__global__ __launch_bounds__(256) void k_qproj(
    const float* __restrict__ h, const float* __restrict__ qw,
    const float* __restrict__ qb, float* __restrict__ q)
{
  __shared__ float sh[H_SZ];
  const int bb = blockIdx.y;
  const float* hl = h + ((size_t)bb * T_SZ + (T_SZ - 1)) * H_SZ;
  for (int c = threadIdx.x; c < H_SZ; c += 256) sh[c] = hl[c];
  __syncthreads();
  const int j = blockIdx.x * 256 + threadIdx.x;
  float accv = qb[j];
  for (int i = 0; i < H_SZ; ++i) accv += sh[i] * qw[(size_t)i * H_SZ + j];
  q[(size_t)bb * H_SZ + j] = accv;
}

// ----------------------------------------- 8-slot cross-attention read head
__global__ __launch_bounds__(256) void k_readhead(
    const float* __restrict__ h, const float* __restrict__ q,
    const int* __restrict__ idx_all, float* __restrict__ ctx)
{
  __shared__ float sq[H_SZ];
  __shared__ float sbuf[256];
  __shared__ float attn[SLOTS];
  __shared__ int sl[SLOTS];
  const int tid = threadIdx.x;
  const int bb = blockIdx.x;
  for (int c = tid; c < H_SZ; c += 256) sq[c] = q[(size_t)bb * H_SZ + c];
  if (tid < SLOTS) sl[tid] = idx_all[bb * SLOTS + tid];
  __syncthreads();
  float part[SLOTS];
#pragma unroll
  for (int s = 0; s < SLOTS; ++s) part[s] = 0.f;
  for (int c = tid; c < H_SZ; c += 256) {
    const float qq = sq[c];
#pragma unroll
    for (int s = 0; s < SLOTS; ++s)
      part[s] += h[((size_t)bb * T_SZ + sl[s]) * H_SZ + c] * qq;
  }
#pragma unroll
  for (int s = 0; s < SLOTS; ++s) {
    float sc = blk_sum(part[s], sbuf);
    if (tid == 0) attn[s] = sc;
  }
  __syncthreads();
  if (tid == 0) {
    float mx = attn[0];
    for (int s = 1; s < SLOTS; ++s) mx = fmaxf(mx, attn[s]);
    float se = 0.f;
    for (int s = 0; s < SLOTS; ++s) { attn[s] = expf(attn[s] - mx); se += attn[s]; }
    float inv = 1.0f / se;
    for (int s = 0; s < SLOTS; ++s) attn[s] *= inv;
  }
  __syncthreads();
  for (int c = tid; c < H_SZ; c += 256) {
    float accv = 0.f;
#pragma unroll
    for (int s = 0; s < SLOTS; ++s)
      accv += attn[s] * h[((size_t)bb * T_SZ + sl[s]) * H_SZ + c];
    ctx[(size_t)bb * H_SZ + c] = accv;
  }
}

// ------------------------- out = ctx @ out_w + out_b; stream out_w exactly once
__global__ __launch_bounds__(256) void k_outgemm(
    const float* __restrict__ ctx, const float* __restrict__ ow,
    const float* __restrict__ ob, float* __restrict__ out)
{
  __shared__ float sctx[B_SZ * H_SZ];   // 64 KB: all 16 ctx rows
  for (int i = threadIdx.x; i < B_SZ * H_SZ; i += 256) sctx[i] = ctx[i];
  __syncthreads();
  const int col = blockIdx.x * 256 + threadIdx.x;
  float accv[B_SZ];
#pragma unroll
  for (int bb = 0; bb < B_SZ; ++bb) accv[bb] = 0.f;
  for (int i = 0; i < H_SZ; ++i) {
    const float w = ow[(size_t)i * V_SZ + col];
#pragma unroll
    for (int bb = 0; bb < B_SZ; ++bb) accv[bb] += sctx[bb * H_SZ + i] * w;
  }
  const float bias = ob[col];
#pragma unroll
  for (int bb = 0; bb < B_SZ; ++bb) out[(size_t)bb * V_SZ + col] = accv[bb] + bias;
}

// ----------------------------------------------------------------------------
extern "C" void kernel_launch(void* const* d_in, const int* in_sizes, int n_in,
                              void* d_out, int out_size, void* d_ws, size_t ws_size,
                              hipStream_t stream) {
  const int*   seq   = (const int*)  d_in[0];
  const float* embed = (const float*)d_in[1];
  const float* ff_w1 = (const float*)d_in[2];
  const float* ff_b1 = (const float*)d_in[3];
  const float* ff_w2 = (const float*)d_in[4];
  const float* ff_b2 = (const float*)d_in[5];
  const float* ln_g  = (const float*)d_in[6];
  const float* ln_b  = (const float*)d_in[7];
  const float* g1_w  = (const float*)d_in[8];
  const float* g1_b  = (const float*)d_in[9];
  const float* g2_w  = (const float*)d_in[10];
  const float* g2_b  = (const float*)d_in[11];
  const float* q_w   = (const float*)d_in[12];
  const float* q_b   = (const float*)d_in[13];
  const float* out_w = (const float*)d_in[14];
  const float* out_b = (const float*)d_in[15];
  float* out = (float*)d_out;
  (void)in_sizes; (void)n_in; (void)out_size; (void)ws_size;

  char* ws = (char*)d_ws;
  size_t off = 0;
  auto carve = [&](size_t bytes) -> char* {
    char* p = ws + off;
    off += (bytes + 255) & ~(size_t)255;
    return p;
  };
  float*          e_f32   = (float*)         carve((size_t)MT * H_SZ * 4);
  unsigned short* e_bf16  = (unsigned short*)carve((size_t)MT * H_SZ * 2);
  unsigned short* w1t     = (unsigned short*)carve((size_t)H_SZ * N1 * 2);
  unsigned short* w2t     = (unsigned short*)carve((size_t)N1 * H_SZ * 2);
  unsigned short* mid     = (unsigned short*)carve((size_t)MT * N1 * 2);
  float*          hbuf    = (float*)         carve((size_t)MT * H_SZ * 4);
  float*          gate1   = (float*)         carve((size_t)MT * 4);
  float*          dotg2   = (float*)         carve((size_t)MT * 4);
  float*          ctxmean = (float*)         carve((size_t)B_SZ * H_SZ * 4);
  float*          c2      = (float*)         carve((size_t)B_SZ * 4);
  int*            idx_all = (int*)           carve((size_t)B_SZ * SLOTS * 4);
  float*          qv      = (float*)         carve((size_t)B_SZ * H_SZ * 4);
  float*          ctxv    = (float*)         carve((size_t)B_SZ * H_SZ * 4);

  hipLaunchKernelGGL(k_embed_gather, dim3(MT), dim3(256), 0, stream,
                     seq, embed, e_f32, e_bf16);
  hipLaunchKernelGGL(k_transpose_bf16, dim3((H_SZ * N1 + 255) / 256), dim3(256), 0, stream,
                     ff_w1, w1t, H_SZ, N1);
  hipLaunchKernelGGL(k_transpose_bf16, dim3((N1 * H_SZ + 255) / 256), dim3(256), 0, stream,
                     ff_w2, w2t, N1, H_SZ);
  // GEMM1: mid = ReLU(e @ W1 + b1)  -> bf16
  hipLaunchKernelGGL(k_gemm_bf16, dim3(MT / 128, N1 / 128), dim3(256), 0, stream,
                     e_bf16, w1t, ff_b1, (const float*)nullptr, mid, (float*)nullptr,
                     N1, H_SZ, 0);
  // GEMM2: hbuf = mid @ W2 + b2 + e  -> f32
  hipLaunchKernelGGL(k_gemm_bf16, dim3(MT / 128, H_SZ / 128), dim3(256), 0, stream,
                     mid, w2t, ff_b2, (const float*)e_f32, (unsigned short*)nullptr, hbuf,
                     H_SZ, N1, 1);
  hipLaunchKernelGGL(k_ln_gates, dim3(MT), dim3(256), 0, stream,
                     hbuf, ln_g, ln_b, g1_w, g1_b, g2_w, gate1, dotg2);
  hipLaunchKernelGGL(k_select1, dim3(B_SZ), dim3(256), 0, stream,
                     gate1, hbuf, g2_w, ctxmean, c2, idx_all);
  hipLaunchKernelGGL(k_select2, dim3(B_SZ), dim3(256), 0, stream,
                     dotg2, c2, g2_b, idx_all);
  hipLaunchKernelGGL(k_qproj, dim3(H_SZ / 256, B_SZ), dim3(256), 0, stream,
                     hbuf, q_w, q_b, qv);
  hipLaunchKernelGGL(k_readhead, dim3(B_SZ), dim3(256), 0, stream,
                     hbuf, qv, idx_all, ctxv);
  hipLaunchKernelGGL(k_outgemm, dim3(V_SZ / 256), dim3(256), 0, stream,
                     ctxv, out_w, out_b, out);
}
